// ParallelDroplessMoE_12661563588728
// MI455X (gfx1250) — matrix operand order repaired and checked by reading, not compiled, not run on hardware
//
#include <hip/hip_runtime.h>
#include <math.h>

// ---------------------------------------------------------------------------
// Dropless MoE for MI455X (gfx1250, wave32, WMMA + TDM).
//   SL=4096 BS=8 HS=512 FFN=2048 E=8 top_k=2  -> T = 32768 tokens
// Strategy: bf16 WMMA (v_wmma_f32_16x16x32_bf16, f32 accum) for both GEMMs,
// fused gelu + combine-weight scaling, per-32-token-tile expert skipping,
// Tensor Data Mover (tensor_load_to_lds) staging of the X tile into LDS.
// Workspace layout (needs ~65 MB):
//   [0)            xb  : T*HS        bf16 (32 MB)
//   [+16M elems)   w1t : E*FFN*HS    bf16 (16 MB)  (transposed: [e][n][k])
//   [+...  )       w2t : E*HS*FFN    bf16 (16 MB)  (transposed: [e][n][k])
//   [+...  )       comb: T*E         f32  ( 1 MB)
// ---------------------------------------------------------------------------

#define SLEN   4096
#define BSZ    8
#define HSD    512
#define FFND   2048
#define NE     8
#define TTOK   (SLEN * BSZ)      // 32768
#define TILE_T 32                // tokens per block
#define CHUNK  64                // FFN slice per inner iteration

typedef unsigned int   u32;
typedef unsigned short u16;
typedef unsigned long long u64;
typedef __attribute__((ext_vector_type(16))) __bf16 v16bf;
typedef __attribute__((ext_vector_type(8)))  float  v8f;
typedef __attribute__((ext_vector_type(4)))  u32    u32x4;
typedef __attribute__((ext_vector_type(4)))  u32    uint32x4;
typedef __attribute__((ext_vector_type(8)))  int    int32x8;
typedef __attribute__((ext_vector_type(4)))  int    int32x4;

union Frag { v16bf v; u32x4 q[2]; };

__device__ __forceinline__ u16 f2bf(float f) {
  u32 u = __float_as_uint(f);
  return (u16)((u + 0x7FFFu + ((u >> 16) & 1u)) >> 16);   // RNE
}

// ---------------------------------------------------------------- converters
__global__ __launch_bounds__(256) void k_cvt_x(const float* __restrict__ s,
                                               u16* __restrict__ d, int n) {
  for (int i = blockIdx.x * 256 + threadIdx.x; i < n; i += gridDim.x * 256)
    d[i] = f2bf(s[i]);
}

// w1: [E][HS][FFN] fp32  ->  w1t: [E][FFN][HS] bf16 (K-contiguous B rows)
__global__ __launch_bounds__(256) void k_cvt_w1(const float* __restrict__ s,
                                                u16* __restrict__ d) {
  const int n = NE * HSD * FFND;
  for (int i = blockIdx.x * 256 + threadIdx.x; i < n; i += gridDim.x * 256) {
    int e = i / (HSD * FFND), rem = i % (HSD * FFND);
    int k = rem / FFND, c = rem % FFND;
    d[(size_t)e * FFND * HSD + (size_t)c * HSD + k] = f2bf(s[i]);
  }
}

// w2: [E][FFN][HS] fp32  ->  w2t: [E][HS][FFN] bf16 (K-contiguous B rows)
__global__ __launch_bounds__(256) void k_cvt_w2(const float* __restrict__ s,
                                                u16* __restrict__ d) {
  const int n = NE * FFND * HSD;
  for (int i = blockIdx.x * 256 + threadIdx.x; i < n; i += gridDim.x * 256) {
    int e = i / (FFND * HSD), rem = i % (FFND * HSD);
    int k = rem / HSD, c = rem % HSD;
    d[(size_t)e * HSD * FFND + (size_t)c * FFND + k] = f2bf(s[i]);
  }
}

// ---------------------------------------------------------------- router
__global__ __launch_bounds__(256) void k_router(const float* __restrict__ x,
                                                const float* __restrict__ wr,
                                                float* __restrict__ comb) {
  const int lane = threadIdx.x & 31;
  const int wave = threadIdx.x >> 5;
  const int t = blockIdx.x * 8 + wave;

  float acc[NE];
#pragma unroll
  for (int e = 0; e < NE; e++) acc[e] = 0.0f;

  const float* xr = x + (size_t)t * HSD;
  for (int h = lane; h < HSD; h += 32) {
    float xv = xr[h];
    const float* w = wr + (size_t)h * NE;
#pragma unroll
    for (int e = 0; e < NE; e++) acc[e] = fmaf(xv, w[e], acc[e]);
  }
#pragma unroll
  for (int off = 16; off > 0; off >>= 1) {
#pragma unroll
    for (int e = 0; e < NE; e++) acc[e] += __shfl_xor(acc[e], off, 32);
  }

  if (lane == 0) {
    float mx = acc[0];
#pragma unroll
    for (int e = 1; e < NE; e++) mx = fmaxf(mx, acc[e]);
    float sc[NE], s = 0.0f;
#pragma unroll
    for (int e = 0; e < NE; e++) { sc[e] = __expf(acc[e] - mx); s += sc[e]; }
    float inv = 1.0f / s;
#pragma unroll
    for (int e = 0; e < NE; e++) sc[e] *= inv;

    const bool dense = (t / BSZ) < (SLEN / NE);   // first trim_idx seq positions
    float outw[NE];
    if (dense) {
#pragma unroll
      for (int e = 0; e < NE; e++) outw[e] = sc[e];
    } else {
      int i1 = 0;
#pragma unroll
      for (int e = 1; e < NE; e++) if (sc[e] > sc[i1]) i1 = e;
      int i2 = (i1 == 0) ? 1 : 0;
#pragma unroll
      for (int e = 0; e < NE; e++) if (e != i1 && sc[e] > sc[i2]) i2 = e;
#pragma unroll
      for (int e = 0; e < NE; e++) outw[e] = (e == i1 || e == i2) ? sc[e] : 0.0f;
    }
    float* c = comb + (size_t)t * NE;
#pragma unroll
    for (int e = 0; e < NE; e++) c[e] = outw[e];
  }
}

// ---------------------------------------------------------------- MoE MLP
__global__ __launch_bounds__(256) void k_moe(const u16* __restrict__ xb,
                                             const u16* __restrict__ w1t,
                                             const u16* __restrict__ w2t,
                                             const float* __restrict__ comb,
                                             float* __restrict__ out) {
  __shared__ alignas(16) u16   Xs[TILE_T][HSD + 8];     // 33,280 B (padded rows)
  __shared__ alignas(16) u16   Hs[TILE_T][CHUNK + 8];   //  4,608 B
  __shared__ alignas(16) float Cs[TILE_T][NE];          //  1,024 B
  __shared__ int flg[NE];

  const int tid  = threadIdx.x;
  const int lane = tid & 31;
  const int wave = tid >> 5;
  const int hl   = lane >> 4;      // half-lane (ISA fragment layout)
  const int ln   = lane & 15;
  const int t0   = blockIdx.x * TILE_T;

  // ---- stage X tile (bf16) into LDS ------------------------------------
#if __has_builtin(__builtin_amdgcn_tensor_load_to_lds) && \
    __has_builtin(__builtin_amdgcn_s_wait_tensorcnt)
  // Tensor Data Mover: 2-D tile [32 rows x 512 bf16] of tensor [32768 x 512],
  // LDS padding 4 DWORDs after every 256 DWORDs reproduces Xs[32][512+8].
  if (wave == 0) {
    const u64 ga = (u64)(uintptr_t)(xb + (size_t)t0 * HSD);
    const u32 la = (u32)(uintptr_t)&Xs[0][0];   // low 32 bits = LDS offset
    uint32x4 g0;
    g0[0] = 1u;                                  // count=1 valid descriptor
    g0[1] = la;                                  // lds_addr
    g0[2] = (u32)ga;                             // global_addr[31:0]
    g0[3] = (u32)(ga >> 32) | 0x80000000u;       // global_addr[56:32] | type=2
    int32x8 g1;
    g1[0] = (int)((3u << 25) | (7u << 22) | (1u << 20) | (1u << 16));
           // pad_amount=4dw, pad_interval=256dw, pad_en, data_size=2B
    g1[1] = (int)(((u32)HSD & 0xFFFFu) << 16);   // tensor_dim0 = 512 (lo16)
    g1[2] = (int)(((u32)TTOK & 0xFFFFu) << 16);  // dim0 hi=0 | tensor_dim1 lo16
    g1[3] = (int)(((u32)HSD) << 16);             // dim1 hi=0 | tile_dim0 = 512
    g1[4] = TILE_T;                              // tile_dim1 = 32 | tile_dim2=0
    g1[5] = HSD;                                 // tensor_dim0_stride = 512
    g1[6] = 0;
    g1[7] = 0;
    int32x4 z4 = {0, 0, 0, 0};                   // groups 2/3 unused (2-D)
    int32x8 z8 = {0, 0, 0, 0, 0, 0, 0, 0};       // extra group (clang-23 form)
    __builtin_amdgcn_tensor_load_to_lds(g0, g1, z4, z4, z8, 0);
    __builtin_amdgcn_s_wait_tensorcnt(0);        // TENSORcnt -> 0
  }
#else
  for (int idx = tid; idx < TILE_T * (HSD / 8); idx += 256) {
    int row = idx >> 6, c8 = idx & 63;
    u32x4 v = *(const u32x4*)(xb + (size_t)(t0 + row) * HSD + c8 * 8);
    *(u32x4*)(&Xs[row][c8 * 8]) = v;
  }
#endif
  // stage combine weights
  for (int idx = tid; idx < TILE_T * NE; idx += 256)
    Cs[idx >> 3][idx & 7] = comb[(size_t)t0 * NE + idx];
  __syncthreads();
  if (tid < NE) {                  // per-tile expert-active flags
    int any = 0;
    for (int r = 0; r < TILE_T; r++) any |= (Cs[r][tid] != 0.0f);
    flg[tid] = any;
  }
  __syncthreads();

  // phase-2 ownership: token strip (wave&1), 128-col quarter (wave>>1)
  const int tm2 = wave & 1;
  const int nq  = wave >> 1;
  // phase-1 ownership: one 16x16 H tile per wave
  const int tm1 = wave >> 2;       // token strip 0..1
  const int tn1 = wave & 3;        // ffn 16-col group 0..3

  v8f acc[8];
#pragma unroll
  for (int j = 0; j < 8; j++)
#pragma unroll
    for (int r = 0; r < 8; r++) acc[j][r] = 0.0f;

  for (int e = 0; e < NE; e++) {
    if (!flg[e]) continue;         // block-uniform skip
    const u16* W1e = w1t + (size_t)e * FFND * HSD;
    const u16* W2e = w2t + (size_t)e * HSD * FFND;

    for (int ch = 0; ch < FFND / CHUNK; ch++) {
      // ---- phase 1: one 16x16 tile of H per wave ----------------------
      v8f hacc;
#pragma unroll
      for (int r = 0; r < 8; r++) hacc[r] = 0.0f;

      const u16* Ap = &Xs[tm1 * 16 + ln][0];
      const u16* Bp = W1e + (size_t)(ch * CHUNK + tn1 * 16 + ln) * HSD;
#pragma unroll 4
      for (int k0 = 0; k0 < HSD; k0 += 32) {
        Frag a, b;
        a.q[0] = *(const u32x4*)(Ap + k0 + 8 * hl);        // K = 8*hl..+7
        a.q[1] = *(const u32x4*)(Ap + k0 + 16 + 8 * hl);   // K = 16+8*hl..+7
        b.q[0] = *(const u32x4*)(Bp + k0 + 8 * hl);        // K = 8*hl..+7
        b.q[1] = *(const u32x4*)(Bp + k0 + 16 + 8 * hl);
        hacc = __builtin_amdgcn_wmma_f32_16x16x32_bf16(
            false, a.v, false, b.v, (short)0, hacc, false, false);
      }
      // exact gelu, scale by combine weight, store bf16 to LDS
#pragma unroll
      for (int r = 0; r < 8; r++) {
        int tl = tm1 * 16 + hl * 8 + r;                    // M = r + 8*hl
        float v = hacc[r];
        v = 0.5f * v * (1.0f + erff(v * 0.70710678118f));
        v *= Cs[tl][e];
        Hs[tl][tn1 * 16 + ln] = f2bf(v);
      }
      __syncthreads();

      // ---- phase 2: acc += Hs @ W2[chunk, cols] -----------------------
      const u16* Ap2 = &Hs[tm2 * 16 + ln][0];
#pragma unroll
      for (int s = 0; s < 2; s++) {
        Frag a;
        a.q[0] = *(const u32x4*)(Ap2 + s * 32 + 8 * hl);
        a.q[1] = *(const u32x4*)(Ap2 + s * 32 + 16 + 8 * hl);
        const int kg = ch * CHUNK + s * 32;
#pragma unroll
        for (int j = 0; j < 8; j++) {
          Frag b;
          const u16* Bp2 = W2e + (size_t)(nq * 128 + j * 16 + ln) * FFND + kg;
          b.q[0] = *(const u32x4*)(Bp2 + 8 * hl);
          b.q[1] = *(const u32x4*)(Bp2 + 16 + 8 * hl);
          acc[j] = __builtin_amdgcn_wmma_f32_16x16x32_bf16(
              false, a.v, false, b.v, (short)0, acc[j], false, false);
        }
      }
      __syncthreads();             // Hs reused next chunk
    }
  }

  // write 32x512 f32 output tile (each element written exactly once)
#pragma unroll
  for (int j = 0; j < 8; j++) {
    const int col = nq * 128 + j * 16 + ln;
#pragma unroll
    for (int r = 0; r < 8; r++) {
      const int t = t0 + tm2 * 16 + hl * 8 + r;
      out[(size_t)t * HSD + col] = acc[j][r];
    }
  }
}

// ---------------------------------------------------------------- launcher
extern "C" void kernel_launch(void* const* d_in, const int* in_sizes, int n_in,
                              void* d_out, int out_size, void* d_ws, size_t ws_size,
                              hipStream_t stream) {
  (void)in_sizes; (void)n_in; (void)out_size; (void)ws_size;
  const float* x  = (const float*)d_in[0];   // [SL,BS,HS]
  const float* wr = (const float*)d_in[1];   // [HS,E]
  const float* w1 = (const float*)d_in[2];   // [E,HS,FFN]
  const float* w2 = (const float*)d_in[3];   // [E,FFN,HS]
  // d_in[4] = top_k (=2, hardwired in the router's top-2 selection)
  float* out = (float*)d_out;

  u16*  xb   = (u16*)d_ws;
  u16*  w1t  = xb  + (size_t)TTOK * HSD;
  u16*  w2t  = w1t + (size_t)NE * HSD * FFND;
  float* comb = (float*)(w2t + (size_t)NE * FFND * HSD);

  k_cvt_x <<<8192, 256, 0, stream>>>(x, xb, TTOK * HSD);
  k_cvt_w1<<<8192, 256, 0, stream>>>(w1, w1t);
  k_cvt_w2<<<8192, 256, 0, stream>>>(w2, w2t);
  k_router<<<TTOK / 8, 256, 0, stream>>>(x, wr, comb);
  k_moe   <<<TTOK / TILE_T, 256, 0, stream>>>(xb, w1t, w2t, comb, out);
}
